// HamiHeadSymmetry_uuw_multihead_84748294684800
// MI455X (gfx1250) — matrix operand order repaired and checked by reading, not compiled, not run on hardware
//
#include <hip/hip_runtime.h>
#include <hip/hip_bf16.h>
#include <math.h>

// ============================================================================
// HamiHeadSymmetry on MI455X (gfx1250), wave32 + WMMA bf16 16x16x32.
//
//   K0  gen_w3j          : device-side Wigner-3j real-basis table (1225 f32)
//   K1  pack_b  (x20)    : f32 weights -> bf16 packed in WMMA B-fragment order
//   K2  band_bucket      : per-edge distance band, compact edge lists
//   K3  diag_gemm1       : H  = silu(attr @ W1 + b1)           (M=16, WMMA)
//   K4  edge_gemm1       : Hij = silu((attr[d]+attr[s]) @ W1[band] + b1)
//   K5  diag_gemm2exp    : M=32 tile, Wts = H @ W2 + b2 (each B-frag feeds
//                          TWO v_wmma -> halves the L2 weight stream), CG
//                          expansion fully LDS-resident, symmetrize+mask.
//   K6  edge_gemm2exp    : same per band bucket -> out rows [N, N+E)
//
// Staging uses gfx1250 async global->LDS (ASYNCcnt) when available.
// ============================================================================

typedef __attribute__((ext_vector_type(16))) __bf16 v16bf;
typedef __attribute__((ext_vector_type(8)))  float  v8f;

// ---------------- async global->LDS staging (gfx1250 path + fallback) -------
#if __has_builtin(__builtin_amdgcn_global_load_async_to_lds_b128) && \
    __has_builtin(__builtin_amdgcn_s_wait_asynccnt)
#define USE_ASYNC_LDS 1
#else
#define USE_ASYNC_LDS 0
#endif

#if USE_ASYNC_LDS
// Builtin signature (from compiler diagnostic): param0 = int4 __device__* (AS1
// global), param1 = int4 __shared__* (AS3 LDS), then imm offset + imm cpol.
typedef int v4i_vs __attribute__((vector_size(16)));
typedef __attribute__((address_space(1))) v4i_vs gas_v4i;
typedef __attribute__((address_space(3))) v4i_vs las_v4i;
__device__ inline void g2l_b128(const void* g, void* l) {
  // generic->AS1 is numerically a no-op; generic LDS ptr low 32 bits are the
  // workgroup-relative LDS byte offset the instruction expects.
  __builtin_amdgcn_global_load_async_to_lds_b128(
      (gas_v4i*)(unsigned long long)g,
      (las_v4i*)(unsigned)(unsigned long long)l, 0, 0);
}
__device__ inline void async_wait() { __builtin_amdgcn_s_wait_asynccnt(0); }
#else
__device__ inline void g2l_b128(const void* g, void* l) {
  *(uint4*)l = *(const uint4*)g;
}
__device__ inline void async_wait() {}
#endif

// ---------------- instruction metadata (19 CG paths) ----------------
__constant__ int C_lin[19]  = {0,1,2,1,0,1,2,1,2,3,2,1,2,3,0,1,2,3,4};
__constant__ int C_mul1[19] = {3,3,3,2,2,2,2,2,2,2,1,1,1,1,1,1,1,1,1};
__constant__ int C_l1[19]   = {0,0,0,1,1,1,1,1,1,1,2,2,2,2,2,2,2,2,2};
__constant__ int C_o1[19]   = {0,0,0,3,3,3,3,3,3,3,9,9,9,9,9,9,9,9,9};
__constant__ int C_mul2[19] = {3,2,1,3,2,2,2,1,1,1,3,2,2,2,1,1,1,1,1};
__constant__ int C_l2[19]   = {0,1,2,0,1,1,1,2,2,2,0,1,1,1,2,2,2,2,2};
__constant__ int C_o2[19]   = {0,3,9,0,3,3,3,9,9,9,0,3,3,3,9,9,9,9,9};
__constant__ int C_wi[19]   = {0,288,480,576,768,896,1024,1152,1216,1280,
                               1344,1440,1504,1568,1632,1664,1696,1728,1760};
__constant__ int C_bi[19]   = {0,-1,-1,-1,9,-1,-1,-1,-1,-1,-1,-1,-1,-1,13,-1,-1,-1,-1};
__constant__ int C_w3[19]   = {0,1,10,35,44,53,80,125,170,245,350,375,420,495,
                               600,625,700,825,1000};
__constant__ int C_xoff[5]  = {0,32,128,288,512};
#define W3J_TOTAL 1225
#define W3J_PAD   1228   // padded to a b128 multiple for async staging

// ---------------- WMMA fragment helpers (wave32 layouts per ISA 7.12.2) ----
__device__ inline v16bf load_a_frag(const __bf16* A, int lda, int kt, int lane) {
  int r = lane & 15, half = lane >> 4;
  const __bf16* base = A + r * lda + kt * 32 + 8 * half;
  v16bf a;
#pragma unroll
  for (int v = 0; v < 8; ++v) {
    int k0 = 2 * v + ((v >= 4) ? 8 : 0);
    a[2 * v]     = base[k0];
    a[2 * v + 1] = base[k0 + 1];
  }
  return a;
}
__device__ inline v16bf load_b_frag_packed(const __bf16* Bp, int NT, int kt, int nt, int lane) {
  const v16bf* p = (const v16bf*)Bp;
  return p[(kt * NT + nt) * 32 + lane];
}
__device__ inline v8f wmma_bf16(v16bf a, v16bf b, v8f c) {
  return __builtin_amdgcn_wmma_f32_16x16x32_bf16(false, a, false, b, (short)0, c, false, false);
}
__device__ inline float silu(float x) { return x / (1.f + __expf(-x)); }

// ---------------- K1: pack f32 [K][Ncols] -> bf16 B-fragment order ----------
__global__ void pack_b_kernel(const float* __restrict__ src, __bf16* __restrict__ dst,
                              int K, int Ncols, int Npad) {
  int NT = Npad >> 4;
  int total = K * Npad;
  for (int idx = blockIdx.x * blockDim.x + threadIdx.x; idx < total;
       idx += gridDim.x * blockDim.x) {
    int c = idx & 15;
    int lane = (idx >> 4) & 31;
    int tile = idx >> 9;
    int nt = tile % NT, kt = tile / NT;
    int k = kt * 32 + (lane >> 4) * 8 + (c & 7) + (c >> 3) * 16;
    int n = nt * 16 + (lane & 15);
    float v = (k < K && n < Ncols) ? src[k * Ncols + n] : 0.f;
    dst[idx] = (__bf16)v;
  }
}

// ---------------- K0: Wigner 3j real-basis tables (device-generated) --------
struct cplx { double re, im; };
__device__ inline cplx cmulc(cplx a, cplx b) {
  return {a.re * b.re - a.im * b.im, a.re * b.im + a.im * b.re};
}
__device__ double w3j_c(int j1, int j2, int j3, int m1, int m2, int m3, const double* f) {
  if (m1 + m2 + m3 != 0) return 0.0;
  int lo = j1 - j2; if (lo < 0) lo = -lo;
  if (j3 < lo || j3 > j1 + j2) return 0.0;
  double pref = sqrt(f[j1+j2-j3] * f[j1-j2+j3] * f[-j1+j2+j3] / f[j1+j2+j3+1]
                   * f[j1+m1] * f[j1-m1] * f[j2+m2] * f[j2-m2] * f[j3+m3] * f[j3-m3]);
  int t0 = 0;
  if (j2 - j3 - m1 > t0) t0 = j2 - j3 - m1;
  if (j1 - j3 + m2 > t0) t0 = j1 - j3 + m2;
  int t1 = j1 + j2 - j3;
  if (j1 - m1 < t1) t1 = j1 - m1;
  if (j2 + m2 < t1) t1 = j2 + m2;
  double s = 0.0;
  for (int t = t0; t <= t1; ++t) {
    double d = f[t] * f[j3-j2+m1+t] * f[j3-j1-m2+t] * f[j1+j2-j3-t] * f[j1-m1-t] * f[j2+m2-t];
    s += ((t & 1) ? -1.0 : 1.0) / d;
  }
  int p = j1 - j2 - m3;
  return ((p & 1) ? -1.0 : 1.0) * pref * s;
}
__device__ void real_q(int l, cplx* Q, int ld) {
  int d = 2 * l + 1;
  for (int i = 0; i < d; ++i)
    for (int j = 0; j < d; ++j) Q[i * ld + j] = {0.0, 0.0};
  double s2 = 1.0 / sqrt(2.0);
  Q[l * ld + l] = {1.0, 0.0};
  for (int m = 1; m <= l; ++m) {
    double sg = (m & 1) ? -1.0 : 1.0;
    Q[(l + m) * ld + (l + m)] = {sg * s2, 0.0};
    Q[(l + m) * ld + (l - m)] = {s2, 0.0};
    Q[(l - m) * ld + (l - m)] = {0.0, s2};
    Q[(l - m) * ld + (l + m)] = {0.0, -sg * s2};
  }
}
__global__ void gen_w3j_kernel(float* __restrict__ w3j) {
  double fact[16];
  fact[0] = 1.0;
  for (int i = 1; i < 16; ++i) fact[i] = fact[i - 1] * (double)i;
  for (int key = threadIdx.x; key < 19; key += blockDim.x) {
    int l1 = C_l1[key], l2 = C_l2[key], l3 = C_lin[key], off = C_w3[key];
    int d1 = 2 * l1 + 1, d2 = 2 * l2 + 1, d3 = 2 * l3 + 1;
    cplx Q1[25], Q2[25], Q3[81];
    real_q(l1, Q1, 5); real_q(l2, Q2, 5); real_q(l3, Q3, 9);
    double Cc[5][5][9];
    for (int a = 0; a < 5; ++a)
      for (int b = 0; b < 5; ++b)
        for (int c = 0; c < 9; ++c) Cc[a][b][c] = 0.0;
    for (int m1 = -l1; m1 <= l1; ++m1)
      for (int m2 = -l2; m2 <= l2; ++m2) {
        int m3 = -m1 - m2;
        if (m3 >= -l3 && m3 <= l3)
          Cc[l1 + m1][l2 + m2][l3 + m3] = w3j_c(l1, l2, l3, m1, m2, m3, fact);
      }
    int ph = (l1 + l2 + l3) & 3;
    for (int i = 0; i < d1; ++i)
      for (int j = 0; j < d2; ++j)
        for (int k = 0; k < d3; ++k) {
          cplx t = {0.0, 0.0};
          for (int a = 0; a < d1; ++a)
            for (int b = 0; b < d2; ++b) {
              cplx qq = cmulc(Q1[i * 5 + a], Q2[j * 5 + b]);
              for (int c = 0; c < d3; ++c) {
                double cv = Cc[a][b][c];
                if (cv == 0.0) continue;
                cplx q3 = cmulc(qq, Q3[k * 9 + c]);
                t.re += q3.re * cv;
                t.im += q3.im * cv;
              }
            }
          double r = (ph == 0) ? t.re : (ph == 1) ? t.im : (ph == 2) ? -t.re : -t.im;
          w3j[off + (i * d2 + j) * d3 + k] = (float)r;
        }
  }
}

// ---------------- K2: per-edge band + bucket compaction ---------------------
__global__ void band_bucket_kernel(const float* __restrict__ pos, const int* __restrict__ ei,
                                   int* __restrict__ counts, int* __restrict__ lists, int E) {
  int e = blockIdx.x * blockDim.x + threadIdx.x;
  if (e >= E) return;
  int d = ei[e], s = ei[E + e];
  float dx = pos[d * 3 + 0] - pos[s * 3 + 0];
  float dy = pos[d * 3 + 1] - pos[s * 3 + 1];
  float dz = pos[d * 3 + 2] - pos[s * 3 + 2];
  float dist = sqrtf(dx * dx + dy * dy + dz * dz);
  int band = dist < 4.f ? 0 : (dist < 6.f ? 1 : (dist < 9.f ? 2 : 3));
  int p = atomicAdd(&counts[band], 1);
  lists[band * E + p] = e;
}

// ---------------- K3: diag hidden layer (two heads) -------------------------
__global__ __launch_bounds__(256) void diag_gemm1_kernel(
    const float* __restrict__ attr,
    const __bf16* __restrict__ W1p, const float* __restrict__ b1,
    const __bf16* __restrict__ W1bp, const float* __restrict__ b1b,
    __bf16* __restrict__ H, __bf16* __restrict__ Hb, int N) {
  __shared__ __align__(16) __bf16 At[16 * 128];
  int tid = threadIdx.x;
  int rowbase = blockIdx.x * 16;
  for (int idx = tid; idx < 16 * 128; idx += 256) {
    int r = idx >> 7, c = idx & 127;
    int row = rowbase + r;
    At[idx] = (__bf16)(row < N ? attr[row * 128 + c] : 0.f);
  }
  __syncthreads();
  int wave = tid >> 5, lane = tid & 31;
  int n = wave * 16 + (lane & 15);
  int half = lane >> 4;
  for (int which = 0; which < 2; ++which) {
    const __bf16* Wp = which ? W1bp : W1p;
    const float* bias = which ? b1b : b1;
    __bf16* dst = which ? Hb : H;
    v8f acc = {};
#pragma unroll
    for (int kt = 0; kt < 4; ++kt) {
      v16bf a = load_a_frag(At, 128, kt, lane);
      v16bf b = load_b_frag_packed(Wp, 8, kt, wave, lane);
      acc = wmma_bf16(a, b, acc);
    }
    float bv = bias[n];
#pragma unroll
    for (int v = 0; v < 8; ++v) {
      int row = rowbase + v + 8 * half;
      if (row < N) dst[row * 128 + n] = (__bf16)silu(acc[v] + bv);
    }
  }
}

// ---------------- K4: edge hidden layer per band bucket ---------------------
__global__ __launch_bounds__(256) void edge_gemm1_kernel(
    const float* __restrict__ attr, const int* __restrict__ ei,
    const __bf16* __restrict__ W1p_all, const float* __restrict__ b1_all,
    const __bf16* __restrict__ W1bp_all, const float* __restrict__ b1b_all,
    const int* __restrict__ lists, const int* __restrict__ counts,
    __bf16* __restrict__ H, __bf16* __restrict__ Hb, int E) {
  int band = blockIdx.y;
  int cnt = counts[band];
  int base = blockIdx.x * 16;
  if (base >= cnt) return;
  __shared__ __align__(16) __bf16 At[16 * 128];
  __shared__ int elist[16];
  int tid = threadIdx.x;
  if (tid < 16) {
    int p = base + tid;
    elist[tid] = (p < cnt) ? lists[band * E + p] : -1;
  }
  __syncthreads();
  for (int idx = tid; idx < 16 * 128; idx += 256) {
    int r = idx >> 7, c = idx & 127;
    int e = elist[r];
    float v = 0.f;
    if (e >= 0) v = attr[ei[e] * 128 + c] + attr[ei[E + e] * 128 + c];
    At[idx] = (__bf16)v;
  }
  __syncthreads();
  int wave = tid >> 5, lane = tid & 31;
  int n = wave * 16 + (lane & 15);
  int half = lane >> 4;
  const __bf16* W1p  = W1p_all  + band * (128 * 128);
  const __bf16* W1bp = W1bp_all + band * (128 * 128);
  const float* b1  = b1_all  + band * 128;
  const float* b1b = b1b_all + band * 128;
  for (int which = 0; which < 2; ++which) {
    const __bf16* Wp = which ? W1bp : W1p;
    const float* bias = which ? b1b : b1;
    __bf16* dst = which ? Hb : H;
    v8f acc = {};
#pragma unroll
    for (int kt = 0; kt < 4; ++kt) {
      v16bf a = load_a_frag(At, 128, kt, lane);
      v16bf b = load_b_frag_packed(Wp, 8, kt, wave, lane);
      acc = wmma_bf16(a, b, acc);
    }
    float bv = bias[n];
#pragma unroll
    for (int v = 0; v < 8; ++v) {
      int e = elist[v + 8 * half];
      if (e >= 0) dst[e * 128 + n] = (__bf16)silu(acc[v] + bv);
    }
  }
}

// ---------------- expansion body over 32 LDS-resident rows ------------------
__device__ void expansion_body32(const __bf16* Wts, const float* Xt, const float* Bt,
                                 const float* W3s, float* Ot, int tid) {
  for (int t = 0; t < 19; ++t) {
    int lin = C_lin[t], din = 2 * lin + 1;
    int m1 = C_mul1[t], m2 = C_mul2[t];
    int d1 = 2 * C_l1[t] + 1, d2 = 2 * C_l2[t] + 1;
    int o1 = C_o1[t], o2 = C_o2[t];
    int wioff = C_wi[t], bioff = C_bi[t], w3off = C_w3[t];
    int xoff = C_xoff[lin];
    int stride_w = m1 * m2;
    int jobs = 32 * m1 * m2 * din;
    for (int idx = tid; idx < jobs; idx += 256) {
      int k = idx % din; int rest = idx / din;
      int v2 = rest % m2; rest /= m2;
      int u = rest % m1; int n = rest / m1;
      const __bf16* wp = Wts + n * 1792 + wioff + u * m2 + v2;
      const float* xp = Xt + n * 800 + xoff + k;
      float s = 0.f;
#pragma unroll 8
      for (int w = 0; w < 32; ++w) s += (float)wp[w * stride_w] * xp[w * din];
      if (bioff >= 0) s += Bt[n * 16 + bioff + u * m2 + v2];
      s *= (1.f / 32.f);
      const float* cg = W3s + w3off;
      for (int i = 0; i < d1; ++i)
        for (int jj = 0; jj < d2; ++jj) {
          float c = cg[(i * d2 + jj) * din + k];
          if (c != 0.f)
            atomicAdd(&Ot[n * 196 + (o1 + u * d1 + i) * 14 + (o2 + v2 * d2 + jj)], c * s);
        }
    }
  }
}

// ---------------- K5: diag GEMM2 (M=32, shared B-frags) + expansion ---------
__global__ __launch_bounds__(256) void diag_gemm2exp_kernel(
    const __bf16* __restrict__ H, const __bf16* __restrict__ Hb,
    const __bf16* __restrict__ W2p, const float* __restrict__ b2,
    const __bf16* __restrict__ W2bp, const float* __restrict__ b2b,
    const float* __restrict__ xsrc, const float* __restrict__ mask,
    const float* __restrict__ w3j, float* __restrict__ out, int N) {
  __shared__ __align__(16) __bf16 Ht[32 * 128];
  __shared__ __align__(16) __bf16 Hbt[32 * 128];
  __shared__ __align__(16) __bf16 Wts[32 * 1792];
  __shared__ __align__(16) float Xt[32 * 800];
  __shared__ float Bt[32 * 16];
  __shared__ float Ot[32 * 196];
  __shared__ __align__(16) float W3s[W3J_PAD];
  int tid = threadIdx.x;
  int rowbase = blockIdx.x * 32;
  // ---- async staging: Ht/Hbt (512 b128 chunks each), Xt (6400), W3s (307)
  for (int idx = tid; idx < 512; idx += 256) {
    int r = idx >> 4, c = idx & 15;
    int row = rowbase + r;
    if (row < N) {
      g2l_b128(H  + row * 128 + c * 8, Ht  + idx * 8);
      g2l_b128(Hb + row * 128 + c * 8, Hbt + idx * 8);
    } else {
#pragma unroll
      for (int q = 0; q < 8; ++q) { Ht[idx*8+q] = (__bf16)0.f; Hbt[idx*8+q] = (__bf16)0.f; }
    }
  }
  for (int idx = tid; idx < 6400; idx += 256) {
    int r = idx / 200, q = idx % 200;
    int row = rowbase + r;
    if (row < N) g2l_b128(xsrc + row * 800 + q * 4, Xt + r * 800 + q * 4);
    else {
#pragma unroll
      for (int z = 0; z < 4; ++z) Xt[r * 800 + q * 4 + z] = 0.f;
    }
  }
  for (int idx = tid; idx < 307; idx += 256) g2l_b128(w3j + idx * 4, W3s + idx * 4);
  for (int idx = tid; idx < 32 * 196; idx += 256) Ot[idx] = 0.f;
  async_wait();
  __syncthreads();
  int wave = tid >> 5, lane = tid & 31;
  int n16 = lane & 15, half = lane >> 4;
  // ---- preload A fragments for both 16-row sub-tiles (register-resident)
  v16bf a0[4], a1[4];
#pragma unroll
  for (int kt = 0; kt < 4; ++kt) {
    a0[kt] = load_a_frag(Ht, 128, kt, lane);
    a1[kt] = load_a_frag(Ht + 16 * 128, 128, kt, lane);
  }
  for (int j = 0; j < 14; ++j) {
    int nt = wave + 8 * j;
    if (j + 1 < 14)
      __builtin_prefetch((const char*)W2p + (size_t)(nt + 8) * 32 * 16 * 2, 0, 0);
    v8f acc0 = {}, acc1 = {};
#pragma unroll
    for (int kt = 0; kt < 4; ++kt) {
      v16bf b = load_b_frag_packed(W2p, 112, kt, nt, lane);
      acc0 = wmma_bf16(a0[kt], b, acc0);   // one B fragment, two WMMAs:
      acc1 = wmma_bf16(a1[kt], b, acc1);   // halves the L2 weight stream
    }
    int col = nt * 16 + n16;
    float bv = b2[col];
#pragma unroll
    for (int v = 0; v < 8; ++v) {
      Wts[(v + 8 * half) * 1792 + col]        = (__bf16)(acc0[v] + bv);
      Wts[(16 + v + 8 * half) * 1792 + col]   = (__bf16)(acc1[v] + bv);
    }
  }
  if (wave < 2) {    // bias head: wave0 -> rows 0-15, wave1 -> rows 16-31
    const __bf16* Hbase = Hbt + wave * 16 * 128;
    v8f acc = {};
#pragma unroll
    for (int kt = 0; kt < 4; ++kt) {
      v16bf a = load_a_frag(Hbase, 128, kt, lane);
      v16bf b = load_b_frag_packed(W2bp, 1, kt, 0, lane);
      acc = wmma_bf16(a, b, acc);
    }
    float bv = (n16 < 14) ? b2b[n16] : 0.f;
#pragma unroll
    for (int v = 0; v < 8; ++v) Bt[(wave * 16 + v + 8 * half) * 16 + n16] = acc[v] + bv;
  }
  __syncthreads();
  expansion_body32(Wts, Xt, Bt, W3s, Ot, tid);
  __syncthreads();
  for (int idx = tid; idx < 32 * 196; idx += 256) {
    int n = idx / 196, r = idx % 196;
    int i = r / 14, jj = r % 14;
    int row = rowbase + n;
    if (row < N) {
      float v = 0.5f * (Ot[n * 196 + i * 14 + jj] + Ot[n * 196 + jj * 14 + i]);
      out[row * 196 + r] = v * mask[row * 196 + r];
    }
  }
}

// ---------------- K6: edge GEMM2 (M=32) + expansion per band bucket ---------
__global__ __launch_bounds__(256) void edge_gemm2exp_kernel(
    const __bf16* __restrict__ H, const __bf16* __restrict__ Hb,
    const __bf16* __restrict__ W2p_all, const float* __restrict__ b2_all,
    const __bf16* __restrict__ W2bp_all, const float* __restrict__ b2b_all,
    const float* __restrict__ xsrc, const float* __restrict__ mask,
    const float* __restrict__ w3j, const int* __restrict__ lists,
    const int* __restrict__ counts, float* __restrict__ out, int N, int E) {
  int band = blockIdx.y;
  int cnt = counts[band];
  int base = blockIdx.x * 32;
  if (base >= cnt) return;
  __shared__ __align__(16) __bf16 Ht[32 * 128];
  __shared__ __align__(16) __bf16 Hbt[32 * 128];
  __shared__ __align__(16) __bf16 Wts[32 * 1792];
  __shared__ __align__(16) float Xt[32 * 800];
  __shared__ float Bt[32 * 16];
  __shared__ float Ot[32 * 196];
  __shared__ __align__(16) float W3s[W3J_PAD];
  __shared__ int elist[32];
  int tid = threadIdx.x;
  if (tid < 32) {
    int p = base + tid;
    elist[tid] = (p < cnt) ? lists[band * E + p] : -1;
  }
  __syncthreads();
  for (int idx = tid; idx < 512; idx += 256) {
    int r = idx >> 4, c = idx & 15;
    int e = elist[r];
    if (e >= 0) {
      g2l_b128(H  + e * 128 + c * 8, Ht  + idx * 8);
      g2l_b128(Hb + e * 128 + c * 8, Hbt + idx * 8);
    } else {
#pragma unroll
      for (int q = 0; q < 8; ++q) { Ht[idx*8+q] = (__bf16)0.f; Hbt[idx*8+q] = (__bf16)0.f; }
    }
  }
  for (int idx = tid; idx < 6400; idx += 256) {
    int r = idx / 200, q = idx % 200;
    int e = elist[r];
    if (e >= 0) g2l_b128(xsrc + (size_t)e * 800 + q * 4, Xt + r * 800 + q * 4);
    else {
#pragma unroll
      for (int z = 0; z < 4; ++z) Xt[r * 800 + q * 4 + z] = 0.f;
    }
  }
  for (int idx = tid; idx < 307; idx += 256) g2l_b128(w3j + idx * 4, W3s + idx * 4);
  for (int idx = tid; idx < 32 * 196; idx += 256) Ot[idx] = 0.f;
  async_wait();
  __syncthreads();
  const __bf16* W2p  = W2p_all  + (size_t)band * (128 * 1792);
  const __bf16* W2bp = W2bp_all + (size_t)band * (128 * 16);
  const float* b2  = b2_all  + band * 1792;
  const float* b2b = b2b_all + band * 14;
  int wave = tid >> 5, lane = tid & 31;
  int n16 = lane & 15, half = lane >> 4;
  v16bf a0[4], a1[4];
#pragma unroll
  for (int kt = 0; kt < 4; ++kt) {
    a0[kt] = load_a_frag(Ht, 128, kt, lane);
    a1[kt] = load_a_frag(Ht + 16 * 128, 128, kt, lane);
  }
  for (int j = 0; j < 14; ++j) {
    int nt = wave + 8 * j;
    if (j + 1 < 14)
      __builtin_prefetch((const char*)W2p + (size_t)(nt + 8) * 32 * 16 * 2, 0, 0);
    v8f acc0 = {}, acc1 = {};
#pragma unroll
    for (int kt = 0; kt < 4; ++kt) {
      v16bf b = load_b_frag_packed(W2p, 112, kt, nt, lane);
      acc0 = wmma_bf16(a0[kt], b, acc0);
      acc1 = wmma_bf16(a1[kt], b, acc1);
    }
    int col = nt * 16 + n16;
    float bv = b2[col];
#pragma unroll
    for (int v = 0; v < 8; ++v) {
      Wts[(v + 8 * half) * 1792 + col]      = (__bf16)(acc0[v] + bv);
      Wts[(16 + v + 8 * half) * 1792 + col] = (__bf16)(acc1[v] + bv);
    }
  }
  if (wave < 2) {
    const __bf16* Hbase = Hbt + wave * 16 * 128;
    v8f acc = {};
#pragma unroll
    for (int kt = 0; kt < 4; ++kt) {
      v16bf a = load_a_frag(Hbase, 128, kt, lane);
      v16bf b = load_b_frag_packed(W2bp, 1, kt, 0, lane);
      acc = wmma_bf16(a, b, acc);
    }
    float bv = (n16 < 14) ? b2b[n16] : 0.f;
#pragma unroll
    for (int v = 0; v < 8; ++v) Bt[(wave * 16 + v + 8 * half) * 16 + n16] = acc[v] + bv;
  }
  __syncthreads();
  expansion_body32(Wts, Xt, Bt, W3s, Ot, tid);
  __syncthreads();
  for (int idx = tid; idx < 32 * 196; idx += 256) {
    int n = idx / 196, r = idx % 196;
    int e = elist[n];
    if (e >= 0)
      out[(size_t)(N + e) * 196 + r] = Ot[idx] * mask[(size_t)e * 196 + r];
  }
}

// ============================================================================
extern "C" void kernel_launch(void* const* d_in, const int* in_sizes, int n_in,
                              void* d_out, int out_size, void* d_ws, size_t ws_size,
                              hipStream_t stream) {
  const float* pos           = (const float*)d_in[0];
  const float* node_attr     = (const float*)d_in[1];
  const float* fii           = (const float*)d_in[2];
  const float* fij           = (const float*)d_in[3];
  const int*   ei            = (const int*)d_in[4];
  const float* diag_mask     = (const float*)d_in[5];
  const float* non_diag_mask = (const float*)d_in[6];
  const float* fii_w1  = (const float*)d_in[7];
  const float* fii_b1  = (const float*)d_in[8];
  const float* fii_w2  = (const float*)d_in[9];
  const float* fii_b2  = (const float*)d_in[10];
  const float* fiib_w1 = (const float*)d_in[11];
  const float* fiib_b1 = (const float*)d_in[12];
  const float* fiib_w2 = (const float*)d_in[13];
  const float* fiib_b2 = (const float*)d_in[14];
  const float* fij_w1  = (const float*)d_in[15];
  const float* fij_b1  = (const float*)d_in[16];
  const float* fij_w2  = (const float*)d_in[17];
  const float* fij_b2  = (const float*)d_in[18];
  const float* fijb_w1 = (const float*)d_in[19];
  const float* fijb_b1 = (const float*)d_in[20];
  const float* fijb_w2 = (const float*)d_in[21];
  const float* fijb_b2 = (const float*)d_in[22];

  const int N = in_sizes[1] / 128;
  const int E = in_sizes[3] / 800;
  float* out = (float*)d_out;

  char* ws = (char*)d_ws;
  size_t cur = 0;
  auto alloc = [&](size_t bytes) -> void* {
    void* p = ws + cur;
    cur = (cur + bytes + 255) & ~(size_t)255;
    return p;
  };
  int*    counts = (int*)alloc(4 * sizeof(int));
  int*    lists  = (int*)alloc((size_t)4 * E * sizeof(int));
  float*  w3j    = (float*)alloc(W3J_PAD * sizeof(float));
  __bf16* pWii1  = (__bf16*)alloc((size_t)128 * 128 * 2);
  __bf16* pWiib1 = (__bf16*)alloc((size_t)128 * 128 * 2);
  __bf16* pWii2  = (__bf16*)alloc((size_t)128 * 1792 * 2);
  __bf16* pWiib2 = (__bf16*)alloc((size_t)128 * 16 * 2);
  __bf16* pWij1  = (__bf16*)alloc((size_t)4 * 128 * 128 * 2);
  __bf16* pWijb1 = (__bf16*)alloc((size_t)4 * 128 * 128 * 2);
  __bf16* pWij2  = (__bf16*)alloc((size_t)4 * 128 * 1792 * 2);
  __bf16* pWijb2 = (__bf16*)alloc((size_t)4 * 128 * 16 * 2);
  __bf16* Hii    = (__bf16*)alloc((size_t)N * 128 * 2);
  __bf16* Hiib   = (__bf16*)alloc((size_t)N * 128 * 2);
  __bf16* Hij    = (__bf16*)alloc((size_t)E * 128 * 2);
  __bf16* Hijb   = (__bf16*)alloc((size_t)E * 128 * 2);
  (void)ws_size; (void)n_in; (void)out_size;

  (void)hipMemsetAsync(counts, 0, 4 * sizeof(int), stream);
  gen_w3j_kernel<<<1, 32, 0, stream>>>(w3j);

  auto pack = [&](const float* s, __bf16* d, int K, int Nc, int Np) {
    int tot = K * Np;
    pack_b_kernel<<<(tot + 255) / 256, 256, 0, stream>>>(s, d, K, Nc, Np);
  };
  pack(fii_w1,  pWii1,  128, 128, 128);
  pack(fiib_w1, pWiib1, 128, 128, 128);
  pack(fii_w2,  pWii2,  128, 1792, 1792);
  pack(fiib_w2, pWiib2, 128, 14, 16);
  for (int b = 0; b < 4; ++b) {
    pack(fij_w1  + (size_t)b * 128 * 128,  pWij1  + (size_t)b * 128 * 128,  128, 128, 128);
    pack(fijb_w1 + (size_t)b * 128 * 128,  pWijb1 + (size_t)b * 128 * 128,  128, 128, 128);
    pack(fij_w2  + (size_t)b * 128 * 1792, pWij2  + (size_t)b * 128 * 1792, 128, 1792, 1792);
    pack(fijb_w2 + (size_t)b * 128 * 14,   pWijb2 + (size_t)b * 128 * 16,   128, 14, 16);
  }

  band_bucket_kernel<<<(E + 255) / 256, 256, 0, stream>>>(pos, ei, counts, lists, E);

  int nTiles16N = (N + 15) / 16;
  int nTiles16E = (E + 15) / 16;
  diag_gemm1_kernel<<<nTiles16N, 256, 0, stream>>>(node_attr, pWii1, fii_b1, pWiib1, fiib_b1,
                                                   Hii, Hiib, N);
  edge_gemm1_kernel<<<dim3(nTiles16E, 4), 256, 0, stream>>>(node_attr, ei, pWij1, fij_b1,
                                                            pWijb1, fijb_b1, lists, counts,
                                                            Hij, Hijb, E);

  int nTiles32N = (N + 31) / 32;
  int nTiles32E = (E + 31) / 32;
  diag_gemm2exp_kernel<<<nTiles32N, 256, 0, stream>>>(Hii, Hiib, pWii2, fii_b2, pWiib2,
                                                      fiib_b2, fii, diag_mask, w3j, out, N);
  edge_gemm2exp_kernel<<<dim3(nTiles32E, 4), 256, 0, stream>>>(Hij, Hijb, pWij2, fij_b2,
                                                               pWijb2, fijb_b2, fij,
                                                               non_diag_mask, w3j, lists,
                                                               counts, out, N, E);
}
